// EdgeAwareGINE_77936476554122
// MI455X (gfx1250) — matrix operand order repaired and checked by reading, not compiled, not run on hardware
//
#include <hip/hip_runtime.h>
#include <stdint.h>

#define DD   512
#define EDIM 128

typedef __attribute__((ext_vector_type(8)))  float          v8f;
typedef __attribute__((ext_vector_type(8)))  unsigned short v8us;
typedef __attribute__((ext_vector_type(16))) unsigned short v16us;
typedef __attribute__((ext_vector_type(16))) __bf16         v16bf;

// fragment-native weight layout constants
#define LANE_US 24            // 16 payload halves + 8 pad halves (48B, bank-spread)
#define TILE_US (32 * LANE_US)        // one 16-col x 32-k tile: 768 ushorts
#define KT_US   (32 * TILE_US)        // one 32-k slab, all 32 col-tiles: 24576 ushorts

__device__ __forceinline__ unsigned short f2bf(float f) {
  unsigned int u = __float_as_uint(f);
  u += 0x7FFFu + ((u >> 16) & 1u);          // round-to-nearest-even
  return (unsigned short)(u >> 16);
}

__device__ __forceinline__ v16bf mkfrag(v8us lo, v8us hi) {
  union { v16us us; v16bf bf; } fr;
#pragma unroll
  for (int i = 0; i < 8; ++i) { fr.us[i] = lo[i]; fr.us[i + 8] = hi[i]; }
  return fr.bf;
}

__device__ __forceinline__ unsigned lds_addr_of(const void* p) {
  return (unsigned)(uintptr_t)p;            // low 32 bits of flat-shared addr = LDS addr
}

__device__ __forceinline__ void async_copy_b128(unsigned lds, const void* g) {
  asm volatile("global_load_async_to_lds_b128 %0, %1, off"
               :: "v"(lds), "v"(g) : "memory");
}

__device__ __forceinline__ void wait_async0() {
  asm volatile("s_wait_asynccnt 0x0" ::: "memory");
}

// ---------------------------------------------------------------------------
// One-time: rewrite W[K][512] f32 into bf16 fragment-native layout.
// Wfrag[((kt*32 + ct)*32 + lane)*24 + j] = bf16(W[kt*32 + 16*(lane>>4) + j][ct*16 + (lane&15)])
// ---------------------------------------------------------------------------
__global__ void make_frag_kernel(const float* __restrict__ W,
                                 unsigned short* __restrict__ Wfrag, int K) {
  int idx  = blockIdx.x * blockDim.x + threadIdx.x;   // one thread per (kt,ct,lane)
  int lane = idx & 31;
  int ct   = (idx >> 5) & 31;
  int kt   = idx >> 10;
  if (kt >= K / 32) return;
  int col   = ct * 16 + (lane & 15);
  int kbase = kt * 32 + 16 * (lane >> 4);
  size_t out = ((size_t)(kt * 32 + ct) * 32 + lane) * LANE_US;
#pragma unroll
  for (int j = 0; j < 16; ++j)
    Wfrag[out + j] = f2bf(W[(size_t)(kbase + j) * DD + col]);
#pragma unroll
  for (int j = 16; j < LANE_US; ++j) Wfrag[out + j] = 0;
}

// ---------------------------------------------------------------------------
// Fused GINE edge kernel: msg = relu(edge_attr@We + be + h[src]); agg[dst]+=msg
// Block: 32 edges x 512 cols, 8 waves (2M x 4N), K=128 as 4 async-staged steps.
// ---------------------------------------------------------------------------
__global__ __launch_bounds__(256) void gine_edge_kernel(
    const unsigned short* __restrict__ ea16,   // [E][128] bf16
    const unsigned short* __restrict__ Wef,    // frag layout, 4 kt slabs
    const float* __restrict__ be,
    const float* __restrict__ hIn,
    const long long* __restrict__ srcIdx,
    const long long* __restrict__ dstIdx,
    float* __restrict__ agg,
    int E)
{
  __shared__ unsigned short sA[32 * 136];
  __shared__ unsigned short sB[KT_US];         // 48KB: 32 col-tiles x 32 lanes x 48B
  __shared__ int sSrc[32], sDst[32];

  const int tid  = threadIdx.x;
  const int lane = tid & 31;
  const int wave = tid >> 5;
  const int wm   = wave & 1;
  const int wn   = wave >> 1;
  const int hh   = lane >> 4;
  const int l15  = lane & 15;
  const int e0   = blockIdx.x * 32;
  const unsigned sBaddr = lds_addr_of(&sB[0]);

  if (tid < 32) {
    int e = e0 + tid; if (e >= E) e = E - 1;
    sSrc[tid] = (int)srcIdx[e];
    sDst[tid] = (int)dstIdx[e];
  }
#pragma unroll
  for (int q = 0; q < 2; ++q) {                // stage 32x128 edge_attr tile
    int lin = tid + q * 256;
    int r   = lin >> 4;
    int c0  = (lin & 15) << 3;
    int e = e0 + r; if (e >= E) e = E - 1;
    *(v8us*)(sA + r * 136 + c0) = *(const v8us*)(ea16 + (size_t)e * EDIM + c0);
  }

  v8f acc[8];
#pragma unroll
  for (int nb = 0; nb < 8; ++nb)
#pragma unroll
    for (int i = 0; i < 8; ++i) acc[nb][i] = 0.f;

  for (int kk = 0; kk < 4; ++kk) {
    __syncthreads();                           // prior reads of sB done
    const unsigned short* src = Wef + (size_t)kk * KT_US;
#pragma unroll
    for (int q = 0; q < 12; ++q) {             // 3072 x 16B = 48KB panel
      int c = tid + q * 256;
      async_copy_b128(sBaddr + (unsigned)c * 16, src + (size_t)c * 8);
    }
    wait_async0();
    __syncthreads();

    int r = wm * 16 + l15;
    v8us a0 = *(const v8us*)(sA + r * 136 + kk * 32 + hh * 16);
    v8us a1 = *(const v8us*)(sA + r * 136 + kk * 32 + hh * 16 + 8);
    v16bf afrag = mkfrag(a0, a1);

#pragma unroll
    for (int nb = 0; nb < 8; ++nb) {
      int ct = nb * 4 + wn;
      const unsigned short* bp = sB + ((size_t)ct * 32 + lane) * LANE_US;
      v8us b0 = *(const v8us*)(bp);
      v8us b1 = *(const v8us*)(bp + 8);
      v16bf bfrag = mkfrag(b0, b1);
      acc[nb] = __builtin_amdgcn_wmma_f32_16x16x32_bf16(
          false, afrag, false, bfrag, (short)0, acc[nb], false, false);
    }
  }

#pragma unroll
  for (int nb = 0; nb < 8; ++nb) {
    int col = (nb * 4 + wn) * 16 + l15;
    float bias = be[col];
#pragma unroll
    for (int v = 0; v < 8; ++v) {
      int m = wm * 16 + v + 8 * hh;
      int e = e0 + m;
      if (e < E) {
        int s = sSrc[m], d = sDst[m];
        float val = acc[nb][v] + bias + hIn[(size_t)s * DD + col];
        val = fmaxf(val, 0.f);
        atomicAdd(agg + (size_t)d * DD + col, val);
      }
    }
  }
}

// ---------------------------------------------------------------------------
// C[M,512] = act(A[M,512] @ B + bias); B in frag layout. Double-buffered async.
// Block: 32 rows x 256 cols, 8 waves (2M x 4N), K=512 as 16 k-steps.
// ---------------------------------------------------------------------------
__global__ __launch_bounds__(256) void gemm512_kernel(
    const unsigned short* __restrict__ A16,
    const unsigned short* __restrict__ Bf,     // frag layout, 16 kt slabs
    const float* __restrict__ bias,
    float* __restrict__ outF,
    unsigned short* __restrict__ outB,
    int M, int doRelu)
{
  __shared__ unsigned short sA[2][32 * 40];
  __shared__ unsigned short sB[2][16 * TILE_US];   // 2 x 24KB

  const int tid  = threadIdx.x;
  const int lane = tid & 31;
  const int wave = tid >> 5;
  const int wm   = wave & 1;
  const int wn   = wave >> 1;
  const int hh   = lane >> 4;
  const int l15  = lane & 15;
  const int row0  = blockIdx.x * 32;
  const int nbase = blockIdx.y * 256;
  const int ct0   = blockIdx.y * 16;
  const unsigned sBaddr[2] = { lds_addr_of(&sB[0][0]), lds_addr_of(&sB[1][0]) };

  v8f acc[4];
#pragma unroll
  for (int s = 0; s < 4; ++s)
#pragma unroll
    for (int i = 0; i < 8; ++i) acc[s][i] = 0.f;

  auto stageB = [&](int buf, int kk) {
    const unsigned short* src = Bf + (size_t)kk * KT_US + (size_t)ct0 * TILE_US;
#pragma unroll
    for (int q = 0; q < 6; ++q) {              // 1536 x 16B = 24KB panel
      int c = tid + q * 256;
      async_copy_b128(sBaddr[buf] + (unsigned)c * 16, src + (size_t)c * 8);
    }
  };
  auto stageA = [&](int buf, int kk) {
    if (tid < 128) {
      int r = tid >> 2, c0 = (tid & 3) << 3;
      int rr = row0 + r; if (rr >= M) rr = M - 1;
      *(v8us*)(&sA[buf][r * 40 + c0]) =
          *(const v8us*)(A16 + (size_t)rr * DD + kk * 32 + c0);
    }
  };

  stageB(0, 0);
  stageA(0, 0);

  for (int kk = 0; kk < 16; ++kk) {
    int cur = kk & 1;
    wait_async0();                             // my async copies for buf cur landed
    __syncthreads();                           // everyone's staging done
    if (kk + 1 < 16) {                         // prefetch next panel during compute
      stageB(cur ^ 1, kk + 1);
      stageA(cur ^ 1, kk + 1);
    }

    int r = wm * 16 + l15;
    v8us a0 = *(const v8us*)(&sA[cur][r * 40 + hh * 16]);
    v8us a1 = *(const v8us*)(&sA[cur][r * 40 + hh * 16 + 8]);
    v16bf afrag = mkfrag(a0, a1);

#pragma unroll
    for (int s = 0; s < 4; ++s) {
      int ctl = wn * 4 + s;                    // local col-tile 0..15
      const unsigned short* bp = &sB[cur][((size_t)ctl * 32 + lane) * LANE_US];
      v8us b0 = *(const v8us*)(bp);
      v8us b1 = *(const v8us*)(bp + 8);
      v16bf bfrag = mkfrag(b0, b1);
      acc[s] = __builtin_amdgcn_wmma_f32_16x16x32_bf16(
          false, afrag, false, bfrag, (short)0, acc[s], false, false);
    }
  }

#pragma unroll
  for (int s = 0; s < 4; ++s) {
    int cg = nbase + wn * 64 + s * 16 + l15;
    float bv = bias[cg];
#pragma unroll
    for (int v = 0; v < 8; ++v) {
      int rg = row0 + wm * 16 + v + 8 * hh;
      if (rg < M) {
        float val = acc[s][v] + bv;
        if (doRelu) val = fmaxf(val, 0.f);
        size_t off = (size_t)rg * DD + cg;
        if (outF) outF[off] = val;
        if (outB) outB[off] = f2bf(val);
      }
    }
  }
}

// ---------------------------------------------------------------------------
// Elementwise helpers
// ---------------------------------------------------------------------------
__global__ void cast_bf16_kernel(const float* __restrict__ in,
                                 unsigned short* __restrict__ out, size_t n) {
  size_t i = (size_t)blockIdx.x * blockDim.x + threadIdx.x;
  size_t st = (size_t)gridDim.x * blockDim.x;
  for (; i < n; i += st) out[i] = f2bf(in[i]);
}

__global__ void copy_f32_kernel(const float* __restrict__ in,
                                float* __restrict__ out, size_t n) {
  size_t i = (size_t)blockIdx.x * blockDim.x + threadIdx.x;
  size_t st = (size_t)gridDim.x * blockDim.x;
  for (; i < n; i += st) out[i] = in[i];
}

__global__ void zero_f32_kernel(float* __restrict__ out, size_t n) {
  size_t i = (size_t)blockIdx.x * blockDim.x + threadIdx.x;
  size_t st = (size_t)gridDim.x * blockDim.x;
  for (; i < n; i += st) out[i] = 0.f;
}

__global__ void prep_g_kernel(const float* __restrict__ h,
                              const float* __restrict__ agg,
                              const float* __restrict__ epsArr, int l,
                              unsigned short* __restrict__ g16, size_t n) {
  float sc = 1.0f + epsArr[l];
  size_t i = (size_t)blockIdx.x * blockDim.x + threadIdx.x;
  size_t st = (size_t)gridDim.x * blockDim.x;
  for (; i < n; i += st) g16[i] = f2bf(sc * h[i] + agg[i]);
}

__global__ __launch_bounds__(256) void relu_ln_kernel(
    const float* __restrict__ u, const float* __restrict__ gamma,
    const float* __restrict__ beta, float* __restrict__ hOut,
    unsigned short* __restrict__ h16)
{
  __shared__ float red[256];
  const int row = blockIdx.x, tid = threadIdx.x;
  const float* ur = u + (size_t)row * DD;
  float v0 = fmaxf(ur[tid], 0.f);
  float v1 = fmaxf(ur[tid + 256], 0.f);

  red[tid] = v0 + v1; __syncthreads();
  for (int off = 128; off > 0; off >>= 1) {
    if (tid < off) red[tid] += red[tid + off];
    __syncthreads();
  }
  float mu = red[0] * (1.0f / DD);
  __syncthreads();

  float d0 = v0 - mu, d1 = v1 - mu;
  red[tid] = d0 * d0 + d1 * d1; __syncthreads();
  for (int off = 128; off > 0; off >>= 1) {
    if (tid < off) red[tid] += red[tid + off];
    __syncthreads();
  }
  float inv = rsqrtf(red[0] * (1.0f / DD) + 1e-5f);

  float o0 = d0 * inv * gamma[tid] + beta[tid];
  float o1 = d1 * inv * gamma[tid + 256] + beta[tid + 256];
  size_t base = (size_t)row * DD;
  hOut[base + tid]       = o0;
  hOut[base + tid + 256] = o1;
  h16[base + tid]        = f2bf(o0);
  h16[base + tid + 256]  = f2bf(o1);
}

// ---------------------------------------------------------------------------
extern "C" void kernel_launch(void* const* d_in, const int* in_sizes, int n_in,
                              void* d_out, int out_size, void* d_ws, size_t ws_size,
                              hipStream_t stream)
{
  (void)n_in; (void)out_size; (void)ws_size;
  const float*     x         = (const float*)d_in[0];
  const long long* eidx      = (const long long*)d_in[1];
  const float*     edge_attr = (const float*)d_in[2];
  const float*     We        = (const float*)d_in[3];
  const float*     be        = (const float*)d_in[4];
  const float*     eps       = (const float*)d_in[5];
  const float*     W1        = (const float*)d_in[6];
  const float*     b1        = (const float*)d_in[7];
  const float*     W2        = (const float*)d_in[8];
  const float*     b2        = (const float*)d_in[9];
  const float*     gamma     = (const float*)d_in[10];
  const float*     beta      = (const float*)d_in[11];
  const float*     Wfin      = (const float*)d_in[12];
  const float*     bfv       = (const float*)d_in[13];

  const int Nn = in_sizes[0] / DD;      // 10000
  const int E  = in_sizes[2] / EDIM;    // 160000
  const int L  = 3;

  char* p = (char*)d_ws;
  auto alloc = [&](size_t bytes) -> void* {
    void* r = (void*)p; p += (bytes + 255) & ~(size_t)255; return r;
  };
  float* h   = (float*)alloc((size_t)Nn * DD * 4);
  float* agg = (float*)alloc((size_t)Nn * DD * 4);
  float* u   = (float*)alloc((size_t)Nn * DD * 4);
  unsigned short* g16  = (unsigned short*)alloc((size_t)Nn * DD * 2);
  unsigned short* t16  = (unsigned short*)alloc((size_t)Nn * DD * 2);
  unsigned short* h16  = (unsigned short*)alloc((size_t)Nn * DD * 2);
  unsigned short* ea16 = (unsigned short*)alloc((size_t)E * EDIM * 2);
  // fragment-native weights (incl. 48B/lane padding)
  unsigned short* Wef = (unsigned short*)alloc((size_t)(L * EDIM / 32) * KT_US * 2);
  unsigned short* W1f = (unsigned short*)alloc((size_t)(L * DD / 32) * KT_US * 2);
  unsigned short* W2f = (unsigned short*)alloc((size_t)(L * DD / 32) * KT_US * 2);
  unsigned short* Wff = (unsigned short*)alloc((size_t)(DD / 32) * KT_US * 2);

  auto gsz = [](size_t n) {
    size_t b = (n + 1023) / 1024; if (b > 4096) b = 4096; if (b < 1) b = 1;
    return dim3((unsigned)b);
  };
  dim3 blk(256);

  cast_bf16_kernel<<<gsz((size_t)E * EDIM), blk, 0, stream>>>(edge_attr, ea16, (size_t)E * EDIM);
  // one-time fragment re-layout of all weights (layers stacked along K)
  make_frag_kernel<<<(L * EDIM * 32) / 256, blk, 0, stream>>>(We, Wef, L * EDIM);
  make_frag_kernel<<<(L * DD * 32) / 256, blk, 0, stream>>>(W1, W1f, L * DD);
  make_frag_kernel<<<(L * DD * 32) / 256, blk, 0, stream>>>(W2, W2f, L * DD);
  make_frag_kernel<<<(DD * 32) / 256, blk, 0, stream>>>(Wfin, Wff, DD);
  copy_f32_kernel<<<gsz((size_t)Nn * DD), blk, 0, stream>>>(x, h, (size_t)Nn * DD);

  dim3 ggrid((unsigned)((Nn + 31) / 32), 2);
  const size_t weLayer = (size_t)(EDIM / 32) * KT_US;  // 4 kt slabs per layer
  const size_t wLayer  = (size_t)(DD / 32) * KT_US;    // 16 kt slabs per layer

  for (int l = 0; l < L; ++l) {
    zero_f32_kernel<<<gsz((size_t)Nn * DD), blk, 0, stream>>>(agg, (size_t)Nn * DD);
    gine_edge_kernel<<<(E + 31) / 32, blk, 0, stream>>>(
        ea16, Wef + (size_t)l * weLayer, be + (size_t)l * DD,
        h, eidx, eidx + E, agg, E);
    prep_g_kernel<<<gsz((size_t)Nn * DD), blk, 0, stream>>>(
        h, agg, eps, l, g16, (size_t)Nn * DD);
    gemm512_kernel<<<ggrid, blk, 0, stream>>>(
        g16, W1f + (size_t)l * wLayer, b1 + (size_t)l * DD,
        nullptr, t16, Nn, 1);
    gemm512_kernel<<<ggrid, blk, 0, stream>>>(
        t16, W2f + (size_t)l * wLayer, b2 + (size_t)l * DD,
        u, nullptr, Nn, 0);
    relu_ln_kernel<<<Nn, blk, 0, stream>>>(
        u, gamma + (size_t)l * DD, beta + (size_t)l * DD, h, h16);
  }
  gemm512_kernel<<<ggrid, blk, 0, stream>>>(
      h16, Wff, bfv, (float*)d_out, nullptr, Nn, 0);
}
